// LigerFusedLinearCosineSimilarityLoss_22101901705595
// MI455X (gfx1250) — matrix operand order repaired and checked by reading, not compiled, not run on hardware
//
#include <hip/hip_runtime.h>
#include <hip/hip_bf16.h>

#define BT 2048
#define HDIM 512
#define VDIM 32000
#define VTILES (VDIM / 16)      // 2000
#define ROWS_PER_BLOCK 16
#define NWAVES 8
#define NTHREADS 256
#define NSTAT 6                  // m, l, dot, ss, tt, labLogit

typedef __attribute__((ext_vector_type(16))) __bf16 v16bf;
typedef __attribute__((ext_vector_type(8)))  float  v8f;
typedef __attribute__((ext_vector_type(4)))  unsigned int v4u;

union Frag {
  v16bf v;
  v4u u4[2];
  unsigned short s[16];
};

union Pack8 {
  unsigned short s[8];
  uint4 u;
};

static __device__ __forceinline__ unsigned short f2bfu(float f) {
  // fp32 -> bf16 round-to-nearest-even (bit trick)
  unsigned u = __float_as_uint(f);
  unsigned r = u + 0x7FFFu + ((u >> 16) & 1u);
  return (unsigned short)(r >> 16);
}

// ---- B-fragment loaders: 16 contiguous K-elements of one weight row ----
static __device__ __forceinline__ void loadB(Frag& f, const unsigned short* p) {
  // pre-converted bf16 weights: two 16B loads
  f.u4[0] = *(const v4u*)(p);
  f.u4[1] = *(const v4u*)(p + 8);
}
static __device__ __forceinline__ void loadB(Frag& f, const float* p) {
  // fallback: fp32 weights, convert inline
  const float4* w4 = (const float4*)p;
  #pragma unroll
  for (int q = 0; q < 4; ++q) {
    float4 fs = w4[q];
    f.s[4*q+0]=f2bfu(fs.x); f.s[4*q+1]=f2bfu(fs.y);
    f.s[4*q+2]=f2bfu(fs.z); f.s[4*q+3]=f2bfu(fs.w);
  }
}

// One-time fp32 -> bf16 weight conversion (grid-stride, 8 elems/thread/iter)
__global__ __launch_bounds__(NTHREADS)
void convert_weights(const float* __restrict__ src, unsigned short* __restrict__ dst,
                     size_t n) {
  const size_t stride = (size_t)gridDim.x * blockDim.x * 8;
  for (size_t e = ((size_t)blockIdx.x * blockDim.x + threadIdx.x) * 8; e < n; e += stride) {
    float4 a = *(const float4*)(src + e);
    float4 b = *(const float4*)(src + e + 4);
    Pack8 pk;
    pk.s[0]=f2bfu(a.x); pk.s[1]=f2bfu(a.y); pk.s[2]=f2bfu(a.z); pk.s[3]=f2bfu(a.w);
    pk.s[4]=f2bfu(b.x); pk.s[5]=f2bfu(b.y); pk.s[6]=f2bfu(b.z); pk.s[7]=f2bfu(b.w);
    *(uint4*)(dst + e) = pk.u;
  }
}

template <typename WT>
__global__ __launch_bounds__(NTHREADS)
void fll_cosine_main(const float* __restrict__ SI, const WT* __restrict__ SW,
                     const float* __restrict__ TI, const WT* __restrict__ TW,
                     const int* __restrict__ labels, float* __restrict__ partials) {
  __shared__ unsigned short shAs[ROWS_PER_BLOCK * HDIM];   // 16 KB, student rows bf16
  __shared__ unsigned short shAt[ROWS_PER_BLOCK * HDIM];   // 16 KB, teacher rows bf16
  __shared__ float shRed[NWAVES * 2 * 8 * NSTAT];          // per-(wave,half,slot) stats
  __shared__ int   shLab[ROWS_PER_BLOCK];
  __shared__ float shPart[ROWS_PER_BLOCK];

  const int tid  = threadIdx.x;
  const int lane = tid & 31;
  const int wave = tid >> 5;
  const int rowbase = blockIdx.x * ROWS_PER_BLOCK;

  if (tid < ROWS_PER_BLOCK) shLab[tid] = labels[rowbase + tid];

  // Stage 16 activation rows (student + teacher), fp32 -> bf16, into LDS.
  {
    const float4* s4 = (const float4*)(SI + (size_t)rowbase * HDIM);
    const float4* t4 = (const float4*)(TI + (size_t)rowbase * HDIM);
    for (int i = tid; i < ROWS_PER_BLOCK * HDIM / 4; i += NTHREADS) {
      float4 a = s4[i];
      float4 b = t4[i];
      int o = i * 4;
      shAs[o+0]=f2bfu(a.x); shAs[o+1]=f2bfu(a.y); shAs[o+2]=f2bfu(a.z); shAs[o+3]=f2bfu(a.w);
      shAt[o+0]=f2bfu(b.x); shAt[o+1]=f2bfu(b.y); shAt[o+2]=f2bfu(b.z); shAt[o+3]=f2bfu(b.w);
    }
  }
  __syncthreads();

  const int nIdx = lane & 15;               // column within tile / A row
  const int k0a  = (lane < 16) ? 0 : 16;    // A-fragment K offset (contiguous, matches B)
  const int k0b  = (lane < 16) ? 0 : 16;    // B-fragment K offset (contiguous, matches A)

  int labSlot[8];
  #pragma unroll
  for (int r = 0; r < 8; ++r) labSlot[r] = shLab[r + ((lane >> 4) << 3)];

  // Online softmax + cosine + label-logit accumulators for this lane's 8 C-slots.
  float mrun[8], lrun[8], dacc[8], ssac[8], ttac[8], labAcc[8];
  #pragma unroll
  for (int r = 0; r < 8; ++r) {
    mrun[r] = -1e30f; lrun[r] = 0.f; dacc[r] = 0.f;
    ssac[r] = 0.f;    ttac[r] = 0.f; labAcc[r] = 0.f;
  }

  const unsigned short* pArowS = &shAs[nIdx * HDIM];
  const unsigned short* pArowT = &shAt[nIdx * HDIM];

  for (int j = wave; j < VTILES; j += NWAVES) {
    const int vbase = j * 16;
    const WT* wrS = SW + (size_t)(vbase + nIdx) * HDIM + k0b;
    const WT* wrT = TW + (size_t)(vbase + nIdx) * HDIM + k0b;
    if (j + NWAVES < VTILES) {
      __builtin_prefetch(wrS + (size_t)(NWAVES * 16) * HDIM, 0, 0);
      __builtin_prefetch(wrT + (size_t)(NWAVES * 16) * HDIM, 0, 0);
    }
    v8f accS = {};
    v8f accT = {};
    #pragma unroll 2
    for (int kk = 0; kk < 16; ++kk) {      // K = 512 in steps of 32
      const int kbase = kk * 32;
      Frag fas, fat, fbs, fbt;
      // A fragments from LDS: 16 contiguous K-elements [k0a, k0a+16)
      fas.u4[0] = *(const v4u*)(pArowS + kbase + k0a);
      fas.u4[1] = *(const v4u*)(pArowS + kbase + k0a + 8);
      fat.u4[0] = *(const v4u*)(pArowT + kbase + k0a);
      fat.u4[1] = *(const v4u*)(pArowT + kbase + k0a + 8);
      // B fragments: 16 contiguous K-elements of the weight row
      loadB(fbs, wrS + kbase);
      loadB(fbt, wrT + kbase);
      accS = __builtin_amdgcn_wmma_f32_16x16x32_bf16(false, fas.v, false, fbs.v, (short)0, accS, false, false);
      accT = __builtin_amdgcn_wmma_f32_16x16x32_bf16(false, fat.v, false, fbt.v, (short)0, accT, false, false);
    }
    // Fused epilogue: online softmax + cosine partials, branch-free label capture.
    #pragma unroll
    for (int r = 0; r < 8; ++r) {
      float s = accS[r];
      float t = accT[r];
      dacc[r] = fmaf(s, t, dacc[r]);
      ssac[r] = fmaf(s, s, ssac[r]);
      ttac[r] = fmaf(t, t, ttac[r]);
      float nm = fmaxf(mrun[r], s);
      lrun[r] = lrun[r] * __expf(mrun[r] - nm) + __expf(s - nm);
      mrun[r] = nm;
      labAcc[r] = (labSlot[r] == vbase + nIdx) ? s : labAcc[r];   // v_cndmask, no branch
    }
  }

  // Butterfly-combine across the 16 lanes of each half (masks 1,2,4,8 stay in-half).
  #pragma unroll
  for (int r = 0; r < 8; ++r) {
    #pragma unroll
    for (int mask = 1; mask <= 8; mask <<= 1) {
      float m2 = __shfl_xor(mrun[r], mask, 32);
      float l2 = __shfl_xor(lrun[r], mask, 32);
      float nm = fmaxf(mrun[r], m2);
      lrun[r] = lrun[r] * __expf(mrun[r] - nm) + l2 * __expf(m2 - nm);
      mrun[r] = nm;
      dacc[r]   += __shfl_xor(dacc[r],   mask, 32);
      ssac[r]   += __shfl_xor(ssac[r],   mask, 32);
      ttac[r]   += __shfl_xor(ttac[r],   mask, 32);
      labAcc[r] += __shfl_xor(labAcc[r], mask, 32);  // exactly one nonzero contributor
    }
  }
  if (nIdx == 0) {
    const int halfSel = lane >> 4;
    #pragma unroll
    for (int r = 0; r < 8; ++r) {
      float* p = &shRed[(((wave << 1) | halfSel) * 8 + r) * NSTAT];
      p[0]=mrun[r]; p[1]=lrun[r]; p[2]=dacc[r]; p[3]=ssac[r]; p[4]=ttac[r]; p[5]=labAcc[r];
    }
  }
  __syncthreads();

  // One thread per row: merge the 8 waves, compute per-row loss contribution.
  if (tid < ROWS_PER_BLOCK) {
    const int rr = tid;
    const int halfSel = rr >> 3;
    const int slot = rr & 7;
    float M = -1e30f, L = 0.f, D = 0.f, SS = 0.f, TT = 0.f, LL = 0.f;
    for (int w = 0; w < NWAVES; ++w) {
      const float* p = &shRed[(((w << 1) | halfSel) * 8 + slot) * NSTAT];
      float mi = p[0], li = p[1];
      float nm = fmaxf(M, mi);
      L = L * __expf(M - nm) + li * __expf(mi - nm);
      M = nm;
      D += p[2]; SS += p[3]; TT += p[4]; LL += p[5];
    }
    float nll = (shLab[rr] >= 0) ? (M + __logf(L) - LL) : 0.0f;
    float den = fmaxf(sqrtf(SS), 1e-12f) * fmaxf(sqrtf(TT), 1e-12f);
    float cosv = D / den;
    // 0.5*hard + 0.5*(0.5*(1-cos)) per row, / BT
    shPart[rr] = (0.5f * nll + 0.25f * (1.0f - cosv)) * (1.0f / (float)BT);
  }
  __syncthreads();
  if (tid == 0) {
    float s = 0.f;
    #pragma unroll
    for (int i = 0; i < ROWS_PER_BLOCK; ++i) s += shPart[i];
    partials[blockIdx.x] = s;
  }
}

__global__ void fll_cosine_finalize(const float* __restrict__ partials,
                                    float* __restrict__ out) {
  __shared__ float sh[128];
  const int t = threadIdx.x;
  sh[t] = partials[t];
  __syncthreads();
  for (int off = 64; off > 0; off >>= 1) {
    if (t < off) sh[t] += sh[t + off];
    __syncthreads();
  }
  if (t == 0) out[0] = sh[0];
}

extern "C" void kernel_launch(void* const* d_in, const int* in_sizes, int n_in,
                              void* d_out, int out_size, void* d_ws, size_t ws_size,
                              hipStream_t stream) {
  (void)in_sizes; (void)n_in; (void)out_size;
  const float* SI = (const float*)d_in[0];   // student_input  [BT,H]
  const float* SW = (const float*)d_in[1];   // student_weight [V,H]
  const float* TI = (const float*)d_in[2];   // teacher_input  [BT,H]
  const float* TW = (const float*)d_in[3];   // teacher_weight [V,H]
  const int*   LB = (const int*)d_in[4];     // true_labels    [BT] int32
  float* out = (float*)d_out;
  float* partials = (float*)d_ws;            // first 128 floats of scratch

  const size_t WELEMS = (size_t)VDIM * HDIM;                 // 16.384M per matrix
  const size_t need = 512 + 2 * WELEMS * sizeof(unsigned short);

  dim3 grid(BT / ROWS_PER_BLOCK);            // 128 row-blocks
  dim3 block(NTHREADS);                      // 8 wave32s

  if (ws_size >= need) {
    // Pre-convert both weight matrices to bf16 once; GEMM streams bf16 from L2.
    unsigned short* SWb = (unsigned short*)((char*)d_ws + 512);
    unsigned short* TWb = SWb + WELEMS;
    convert_weights<<<2048, NTHREADS, 0, stream>>>(SW, SWb, WELEMS);
    convert_weights<<<2048, NTHREADS, 0, stream>>>(TW, TWb, WELEMS);
    fll_cosine_main<unsigned short><<<grid, block, 0, stream>>>(SI, SWb, TI, TWb, LB, partials);
  } else {
    // Fallback: convert weights inline in the inner loop.
    fll_cosine_main<float><<<grid, block, 0, stream>>>(SI, SW, TI, TW, LB, partials);
  }
  fll_cosine_finalize<<<1, 128, 0, stream>>>(partials, out);
}
